// SMamba_18622978195837
// MI455X (gfx1250) — matrix operand order repaired and checked by reading, not compiled, not run on hardware
//
#include <hip/hip_runtime.h>
#include <hip/hip_bf16.h>
#include <math.h>

// ---- problem constants (from reference) ----
#define B_     16
#define NFEAT_ 1024     // tokens per batch for the encoder (L)
#define SEQ_   2048     // time length, K of the embed GEMM
#define DM_    128
#define DS_    16
#define DTR_   8
#define NTOK   (B_ * NFEAT_)   // 16384 rows/tokens
#define EPS_   1e-5f

typedef __attribute__((ext_vector_type(16))) __bf16 v16bf;
typedef __attribute__((ext_vector_type(8)))  __bf16 v8bf;
typedef __attribute__((ext_vector_type(8)))  float  v8f;

static __device__ __forceinline__ float sigmoidf_(float x) { return 1.0f / (1.0f + __expf(-x)); }
static __device__ __forceinline__ float siluf_(float x)    { return x * sigmoidf_(x); }

// ---------------------------------------------------------------------------
// Per-(b,feature) mean / stdev / rstd over SEQ samples.
// ---------------------------------------------------------------------------
__global__ void stats_kernel(const float* __restrict__ x,
                             float* __restrict__ mean,
                             float* __restrict__ stdev,
                             float* __restrict__ rstd) {
  int row = blockIdx.x;
  const float* xr = x + (size_t)row * SEQ_;
  float s = 0.f, sq = 0.f;
  for (int i = threadIdx.x; i < SEQ_; i += blockDim.x) {
    float v = xr[i];
    s += v; sq += v * v;
  }
  __shared__ float ss[256], s2[256];
  ss[threadIdx.x] = s; s2[threadIdx.x] = sq;
  __syncthreads();
  for (int off = 128; off > 0; off >>= 1) {
    if ((int)threadIdx.x < off) {
      ss[threadIdx.x] += ss[threadIdx.x + off];
      s2[threadIdx.x] += s2[threadIdx.x + off];
    }
    __syncthreads();
  }
  if (threadIdx.x == 0) {
    float mu  = ss[0] * (1.0f / SEQ_);
    float var = s2[0] * (1.0f / SEQ_) - mu * mu;
    float sd  = sqrtf(var + EPS_);
    mean[row]  = mu;
    stdev[row] = sd;
    rstd[row]  = 1.0f / sd;
  }
}

// ---------------------------------------------------------------------------
// Weight transpose + f32 -> bf16:  W (N x K, row-major)  ->  WT (K x Npad bf16),
// zero padded columns so WMMA tiles past N contribute 0.
// ---------------------------------------------------------------------------
__global__ void wtrans_kernel(const float* __restrict__ W,
                              __bf16* __restrict__ WT,
                              int N, int K, int Npad) {
  int idx = blockIdx.x * blockDim.x + threadIdx.x;
  if (idx >= K * Npad) return;
  int k = idx / Npad;
  int n = idx % Npad;
  float v = (n < N) ? W[(size_t)n * K + k] : 0.0f;
  WT[idx] = (__bf16)v;
}

// ---------------------------------------------------------------------------
// Generic WMMA GEMM:  C[M x Npad] = act( A[M x K] * WT^ + bias )
//   AMODE 0: A is plain f32, row-major, ld = K
//   AMODE 1: A is f32 and each row is normalized on the fly: (a - mean[row]) * rstd[row]
// WT is bf16 K x Npad (already transposed).  One wave32 per 16x16 tile.
// A-matrix VGPR layout (16-bit, 16x32): lane = M%16 + 16*half,
//   elements 0..7  -> K = k0 + 8*half + e,      elements 8..15 -> K = k0 + 16 + 8*half + (e-8)
// B-matrix layout (16-bit, 32x16): lane -> N = ntile*16 + lane%16, element e -> same K as A element e
// C/D layout (f32 16x16): element e -> M = mtile*16 + e + 8*half, N = ntile*16 + lane%16
// ---------------------------------------------------------------------------
template <int AMODE, bool RELU, bool BIAS>
__global__ void gemm_wmma_kernel(const float* __restrict__ A,
                                 const float* __restrict__ mean,
                                 const float* __restrict__ rstd,
                                 const __bf16* __restrict__ BT,
                                 const float* __restrict__ bias,
                                 float* __restrict__ C,
                                 int M, int Npad, int K, int ldc) {
  const int Mt = M >> 4;
  const int Nt = Npad >> 4;
  int tile = blockIdx.x * blockDim.y + threadIdx.y;   // wave-uniform
  if (tile >= Mt * Nt) return;                        // whole wave exits together
  int mt = tile % Mt;
  int nt = tile / Mt;
  int lane = threadIdx.x;
  int mrow = (mt << 4) + (lane & 15);
  int half = lane >> 4;

  float mu = 0.0f, rs = 1.0f;
  if (AMODE == 1) { mu = mean[mrow]; rs = rstd[mrow]; }

  const float* Ar = A + (size_t)mrow * K;
  v8f acc = {0, 0, 0, 0, 0, 0, 0, 0};

  for (int k0 = 0; k0 < K; k0 += 32) {
    v16bf a;
#pragma unroll
    for (int p = 0; p < 4; ++p) {
      float2 u = *(const float2*)(Ar + k0 + 8 * half + 2 * p);
      float2 w = *(const float2*)(Ar + k0 + 16 + 8 * half + 2 * p);
      if (AMODE == 1) {
        u.x = (u.x - mu) * rs; u.y = (u.y - mu) * rs;
        w.x = (w.x - mu) * rs; w.y = (w.y - mu) * rs;
      }
      a[2 * p]     = (__bf16)u.x;
      a[2 * p + 1] = (__bf16)u.y;
      a[8 + 2 * p]     = (__bf16)w.x;
      a[8 + 2 * p + 1] = (__bf16)w.y;
    }
    const __bf16* Br = BT + (size_t)k0 * Npad + (nt << 4) + (lane & 15);
    v16bf b;
#pragma unroll
    for (int i = 0; i < 8; ++i) { b[i] = Br[(size_t)(8 * half + i) * Npad]; b[8 + i] = Br[(size_t)(16 + 8 * half + i) * Npad]; }

    if (k0 + 32 < K) __builtin_prefetch(Ar + k0 + 32, 0, 1);

    acc = __builtin_amdgcn_wmma_f32_16x16x32_bf16(false, a, false, b,
                                                  (short)0, acc, false, false);
  }

  int n = (nt << 4) + (lane & 15);
  float bi = BIAS ? bias[n] : 0.0f;
#pragma unroll
  for (int e = 0; e < 8; ++e) {
    int m = (mt << 4) + e + 8 * half;
    float v = acc[e] + bi;
    if (RELU) v = fmaxf(v, 0.0f);
    C[(size_t)m * ldc + n] = v;
  }
}

// ---------------------------------------------------------------------------
// Depthwise causal conv (DCONV=2) + SiLU.  rev=1 uses the NEXT token as
// "previous" (equivalent to flip -> conv -> flip).
// xz is NTOK x 256; we read its first 128 columns; output xc is NTOK x 128.
// ---------------------------------------------------------------------------
__global__ void conv_silu_kernel(const float* __restrict__ xz,
                                 const float* __restrict__ cw,
                                 const float* __restrict__ cb,
                                 float* __restrict__ xc, int rev) {
  int idx = blockIdx.x * blockDim.x + threadIdx.x;
  if (idx >= NTOK * DM_) return;
  int d   = idx & (DM_ - 1);
  int row = idx >> 7;
  int t   = row & (NFEAT_ - 1);
  float cur  = xz[(size_t)row * (2 * DM_) + d];
  float prev = 0.0f;
  if (!rev) { if (t > 0)          prev = xz[(size_t)(row - 1) * (2 * DM_) + d]; }
  else      { if (t < NFEAT_ - 1) prev = xz[(size_t)(row + 1) * (2 * DM_) + d]; }
  float v = prev * cw[2 * d] + cur * cw[2 * d + 1] + cb[d];
  xc[idx] = siluf_(v);
}

// ---------------------------------------------------------------------------
// dt = softplus(dbc[:, :8] @ dt_w^T + dt_b)   (K=8: plain VALU)
// dbc rows have width 48 (8 dt | 16 B | 16 C | pad? none, 40 real + 8 pad)
// ---------------------------------------------------------------------------
__global__ void dt_kernel(const float* __restrict__ dbc,
                          const float* __restrict__ dtw,
                          const float* __restrict__ dtb,
                          float* __restrict__ dt) {
  int idx = blockIdx.x * blockDim.x + threadIdx.x;
  if (idx >= NTOK * DM_) return;
  int d   = idx & (DM_ - 1);
  int row = idx >> 7;
  const float* r = dbc + (size_t)row * 48;
  float acc = dtb[d];
#pragma unroll
  for (int k = 0; k < DTR_; ++k) acc += r[k] * dtw[d * DTR_ + k];
  float sp = (acc > 20.0f) ? acc : log1pf(__expf(acc));
  dt[idx] = sp;
}

// ---------------------------------------------------------------------------
// Selective scan + D skip + SiLU(z) gating, fused.  One block per batch,
// one thread per channel d; 16 SSM states in registers; shared B/C per step
// staged through LDS.  rev=1 iterates t descending (backward direction).
// ---------------------------------------------------------------------------
__global__ void scan_kernel(const float* __restrict__ dbc,
                            const float* __restrict__ dt,
                            const float* __restrict__ xc,
                            const float* __restrict__ xz,
                            const float* __restrict__ A_log,
                            const float* __restrict__ Dp,
                            float* __restrict__ y, int rev) {
  int b = blockIdx.x;
  int d = threadIdx.x;
  float Aneg[DS_], h[DS_];
#pragma unroll
  for (int s = 0; s < DS_; ++s) {
    Aneg[s] = -__expf(A_log[d * DS_ + s]);
    h[s] = 0.0f;
  }
  float Dd = Dp[d];
  __shared__ float sBC[32];   // [0..15]=B, [16..31]=C for current step

  for (int i = 0; i < NFEAT_; ++i) {
    int t   = rev ? (NFEAT_ - 1 - i) : i;
    int row = b * NFEAT_ + t;
    __syncthreads();
    if (d < 32) sBC[d] = dbc[(size_t)row * 48 + 8 + d];
    __syncthreads();
    float dtv = dt[(size_t)row * DM_ + d];
    float xcv = xc[(size_t)row * DM_ + d];
    float zv  = xz[(size_t)row * (2 * DM_) + DM_ + d];
    float yv = 0.0f;
#pragma unroll
    for (int s = 0; s < DS_; ++s) {
      float dA = __expf(dtv * Aneg[s]);
      h[s] = dA * h[s] + dtv * sBC[s] * xcv;
      yv += h[s] * sBC[16 + s];
    }
    yv += Dd * xcv;
    yv *= siluf_(zv);
    y[(size_t)row * DM_ + d] = yv;
  }
}

// ---------------------------------------------------------------------------
// LayerNorm over DM=128 with up to 2 fused residual adds; one wave32 per token.
// ---------------------------------------------------------------------------
template <int NADD>
__global__ void ln_kernel(const float* __restrict__ a,
                          const float* __restrict__ b,
                          const float* __restrict__ c,
                          const float* __restrict__ g,
                          const float* __restrict__ bta,
                          float* __restrict__ out) {
  int row = blockIdx.x * blockDim.y + threadIdx.y;
  if (row >= NTOK) return;
  int lane = threadIdx.x;
  float v[4];
  float s = 0.f, sq = 0.f;
#pragma unroll
  for (int j = 0; j < 4; ++j) {
    int d = lane + 32 * j;
    float x = a[(size_t)row * DM_ + d];
    if (NADD >= 1) x += b[(size_t)row * DM_ + d];
    if (NADD >= 2) x += c[(size_t)row * DM_ + d];
    v[j] = x; s += x; sq += x * x;
  }
#pragma unroll
  for (int m = 16; m > 0; m >>= 1) {
    s  += __shfl_xor(s,  m, 32);
    sq += __shfl_xor(sq, m, 32);
  }
  float mu   = s  * (1.0f / DM_);
  float var  = sq * (1.0f / DM_) - mu * mu;
  float rstd = rsqrtf(var + EPS_);
#pragma unroll
  for (int j = 0; j < 4; ++j) {
    int d = lane + 32 * j;
    out[(size_t)row * DM_ + d] = (v[j] - mu) * rstd * g[d] + bta[d];
  }
}

// ---------------------------------------------------------------------------
// Head:  dvec[row] = (encF[row,:] . proj_w + proj_b) * stdev[row] + mean[row]
// ---------------------------------------------------------------------------
__global__ void dec_kernel(const float* __restrict__ encF,
                           const float* __restrict__ pw,
                           const float* __restrict__ pb,
                           const float* __restrict__ stdev,
                           const float* __restrict__ mean,
                           float* __restrict__ dvec) {
  int row = blockIdx.x * blockDim.y + threadIdx.y;
  if (row >= NTOK) return;
  int lane = threadIdx.x;
  float s = 0.f;
#pragma unroll
  for (int j = 0; j < 4; ++j) {
    int d = lane + 32 * j;
    s += encF[(size_t)row * DM_ + d] * pw[d];
  }
#pragma unroll
  for (int m = 16; m > 0; m >>= 1) s += __shfl_xor(s, m, 32);
  if (lane == 0) dvec[row] = (s + pb[0]) * stdev[row] + mean[row];
}

// ---------------------------------------------------------------------------
// out[b] = dvec[b,:] . projf_w + projf_b
// ---------------------------------------------------------------------------
__global__ void final_kernel(const float* __restrict__ dvec,
                             const float* __restrict__ pfw,
                             const float* __restrict__ pfb,
                             float* __restrict__ out) {
  int b = blockIdx.x;
  float s = 0.f;
  for (int t = threadIdx.x; t < NFEAT_; t += blockDim.x)
    s += dvec[b * NFEAT_ + t] * pfw[t];
  __shared__ float sm[256];
  sm[threadIdx.x] = s;
  __syncthreads();
  for (int off = 128; off > 0; off >>= 1) {
    if ((int)threadIdx.x < off) sm[threadIdx.x] += sm[threadIdx.x + off];
    __syncthreads();
  }
  if (threadIdx.x == 0) out[b] = sm[0] + pfb[0];
}

// ---------------------------------------------------------------------------
// Host orchestration
// ---------------------------------------------------------------------------
extern "C" void kernel_launch(void* const* d_in, const int* in_sizes, int n_in,
                              void* d_out, int out_size, void* d_ws, size_t ws_size,
                              hipStream_t stream) {
  (void)in_sizes; (void)n_in; (void)out_size; (void)ws_size;

  // Input order: x, embed_w, embed_b, [layer0: m_f(9), m_b(9), c1_w, c1_b,
  // c2_w, c2_b, n1_g, n1_b, n2_g, n2_b] x2 layers, fn_g, fn_b,
  // proj_w, proj_b, projf_w, projf_b   -> 61 tensors total.
  const float* x       = (const float*)d_in[0];
  const float* embed_w = (const float*)d_in[1];
  const float* embed_b = (const float*)d_in[2];
  auto LP = [&](int layer, int off) -> const float* {
    return (const float*)d_in[3 + layer * 26 + off];
  };
  const float* fn_g    = (const float*)d_in[55];
  const float* fn_b    = (const float*)d_in[56];
  const float* proj_w  = (const float*)d_in[57];
  const float* proj_b  = (const float*)d_in[58];
  const float* projf_w = (const float*)d_in[59];
  const float* projf_b = (const float*)d_in[60];

  // ---- workspace carve-out (~92 MB total) ----
  char* wptr = (char*)d_ws;
  auto carve = [&](size_t bytes) -> void* {
    void* p = (void*)wptr;
    wptr += (bytes + 255) & ~(size_t)255;
    return p;
  };
  float*  mean  = (float*)carve((size_t)NTOK * 4);
  float*  stdev = (float*)carve((size_t)NTOK * 4);
  float*  rstd  = (float*)carve((size_t)NTOK * 4);
  float*  enc   = (float*)carve((size_t)NTOK * DM_ * 4);
  float*  enc2  = (float*)carve((size_t)NTOK * DM_ * 4);
  float*  xz    = (float*)carve((size_t)NTOK * 2 * DM_ * 4);
  float*  xcb   = (float*)carve((size_t)NTOK * DM_ * 4);
  float*  dbc   = (float*)carve((size_t)NTOK * 48 * 4);
  float*  dtb   = (float*)carve((size_t)NTOK * DM_ * 4);
  float*  ybuf  = (float*)carve((size_t)NTOK * DM_ * 4);
  float*  fo    = (float*)carve((size_t)NTOK * DM_ * 4);
  float*  bo    = (float*)carve((size_t)NTOK * DM_ * 4);
  float*  ffa   = (float*)carve((size_t)NTOK * DM_ * 4);
  float*  ffb   = (float*)carve((size_t)NTOK * DM_ * 4);
  float*  dvec  = (float*)carve((size_t)NTOK * 4);
  __bf16* wt    = (__bf16*)carve((size_t)SEQ_ * 256 * 2);  // max K*Npad bf16

  const dim3 gemmBlk(32, 8);
  const dim3 waveBlk(32, 8);

  // GEMM launcher: C[M x Npad] = act(A * Wf^T + bias); Wf is (N x K) f32.
  // amode 1 fuses per-row (x - mean) * rstd into the A load.
  auto run_gemm = [&](const float* A, const float* Wf, int N, int K,
                      const float* bias, bool relu, float* C, int amode) {
    int Npad = (N + 15) & ~15;
    int nel  = K * Npad;
    wtrans_kernel<<<(nel + 255) / 256, 256, 0, stream>>>(Wf, wt, N, K, Npad);
    int tiles = (NTOK / 16) * (Npad / 16);
    int grid  = (tiles + 7) / 8;
    if (amode == 1) {
      gemm_wmma_kernel<1, false, true><<<grid, gemmBlk, 0, stream>>>(
          A, mean, rstd, wt, bias, C, NTOK, Npad, K, Npad);
    } else if (relu) {
      gemm_wmma_kernel<0, true, true><<<grid, gemmBlk, 0, stream>>>(
          A, nullptr, nullptr, wt, bias, C, NTOK, Npad, K, Npad);
    } else if (bias) {
      gemm_wmma_kernel<0, false, true><<<grid, gemmBlk, 0, stream>>>(
          A, nullptr, nullptr, wt, bias, C, NTOK, Npad, K, Npad);
    } else {
      gemm_wmma_kernel<0, false, false><<<grid, gemmBlk, 0, stream>>>(
          A, nullptr, nullptr, wt, nullptr, C, NTOK, Npad, K, Npad);
    }
  };

  // 1) normalization statistics over the raw series
  stats_kernel<<<NTOK, 256, 0, stream>>>(x, mean, stdev, rstd);

  // 2) embed GEMM with fused normalization: enc = norm(x) @ embed_w^T + embed_b
  run_gemm(x, embed_w, DM_, SEQ_, embed_b, false, enc, /*amode=*/1);

  const int ewGrid = (NTOK * DM_ + 255) / 256;

  // 3) encoder layers
  for (int ly = 0; ly < 2; ++ly) {
    for (int dir = 0; dir < 2; ++dir) {          // 0 = forward, 1 = backward
      const float* in_w    = LP(ly, dir * 9 + 0);
      const float* conv_w  = LP(ly, dir * 9 + 1);
      const float* conv_b  = LP(ly, dir * 9 + 2);
      const float* xproj_w = LP(ly, dir * 9 + 3);
      const float* dt_w    = LP(ly, dir * 9 + 4);
      const float* dt_b    = LP(ly, dir * 9 + 5);
      const float* A_log   = LP(ly, dir * 9 + 6);
      const float* Dp      = LP(ly, dir * 9 + 7);
      const float* out_w   = LP(ly, dir * 9 + 8);
      float* dirout = dir ? bo : fo;

      run_gemm(enc, in_w, 2 * DM_, DM_, nullptr, false, xz, 0);       // xz
      conv_silu_kernel<<<ewGrid, 256, 0, stream>>>(xz, conv_w, conv_b, xcb, dir);
      run_gemm(xcb, xproj_w, DTR_ + 2 * DS_, DM_, nullptr, false, dbc, 0);
      dt_kernel<<<ewGrid, 256, 0, stream>>>(dbc, dt_w, dt_b, dtb);
      scan_kernel<<<B_, DM_, 0, stream>>>(dbc, dtb, xcb, xz, A_log, Dp, ybuf, dir);
      run_gemm(ybuf, out_w, DM_, DM_, nullptr, false, dirout, 0);     // out proj
    }
    const float* c1_w = LP(ly, 18);
    const float* c1_b = LP(ly, 19);
    const float* c2_w = LP(ly, 20);
    const float* c2_b = LP(ly, 21);
    const float* n1_g = LP(ly, 22);
    const float* n1_b = LP(ly, 23);
    const float* n2_g = LP(ly, 24);
    const float* n2_b = LP(ly, 25);

    // x = LN(enc + fwd + bwd)
    ln_kernel<2><<<NTOK / 8, waveBlk, 0, stream>>>(enc, fo, bo, n1_g, n1_b, enc2);
    // FFN
    run_gemm(enc2, c1_w, DM_, DM_, c1_b, true,  ffa, 0);   // relu(x@c1^T+b)
    run_gemm(ffa,  c2_w, DM_, DM_, c2_b, false, ffb, 0);   // @c2^T+b
    // enc = LN(x + y)
    ln_kernel<1><<<NTOK / 8, waveBlk, 0, stream>>>(enc2, ffb, nullptr, n2_g, n2_b, enc);
  }

  // 4) final LN, projection head, de-normalization, feature reduction
  ln_kernel<0><<<NTOK / 8, waveBlk, 0, stream>>>(enc, nullptr, nullptr, fn_g, fn_b, ffa);
  dec_kernel<<<NTOK / 8, waveBlk, 0, stream>>>(ffa, proj_w, proj_b, stdev, mean, dvec);
  final_kernel<<<B_, 256, 0, stream>>>(dvec, projf_w, projf_b, (float*)d_out);
}
